// AnomalyAttention_67619965108287
// MI455X (gfx1250) — matrix operand order repaired and checked by reading, not compiled, not run on hardware
//
#include <hip/hip_runtime.h>

// ---------------------------------------------------------------------------
// AnomalyAttention for MI455X (gfx1250, wave32, WMMA 16x16x32 f16/f32-acc)
// N=2048, D=1024, H=8, DK=128
// ---------------------------------------------------------------------------

constexpr int kN  = 2048;
constexpr int kD  = 1024;
constexpr int kH  = 8;
constexpr int kDK = 128;

typedef __attribute__((ext_vector_type(16))) _Float16 v16h;
typedef __attribute__((ext_vector_type(8)))  _Float16 v8h;
typedef __attribute__((ext_vector_type(8)))  float    v8f;
typedef __attribute__((ext_vector_type(4)))  float    v4f;

// --------------------------- WMMA helpers ----------------------------------

__device__ __forceinline__ v8f wmma_f16(v16h a, v16h b, v8f c) {
  // D = A(16x32 f16) * B(32x16 f16) + C(16x16 f32)
  return __builtin_amdgcn_wmma_f32_16x16x32_f16(false, a, false, b,
                                                (short)0, c, false, false);
}

// A fragment (16x32, f16) from row-major f16 matrix: lane L holds row (L&15),
// elements k = k0 + 16*half + 0..15
__device__ __forceinline__ v16h load_a_f16(const _Float16* p, int ld, int row0, int k0) {
  int lane = threadIdx.x & 31, m = lane & 15, hf = lane >> 4;
  const _Float16* r = p + (size_t)(row0 + m) * ld + k0;
  v8h lo = *(const v8h*)(r + 16 * hf);
  v8h hi = *(const v8h*)(r + 16 * hf + 8);
  v16h a;
#pragma unroll
  for (int i = 0; i < 8; ++i) { a[i] = lo[i]; a[i + 8] = hi[i]; }
  return a;
}

// Same A fragment but from an f32 source (global x, or LDS scores), converting.
__device__ __forceinline__ v16h load_a_f32cvt(const float* p, int ld, int row0, int k0) {
  int lane = threadIdx.x & 31, m = lane & 15, hf = lane >> 4;
  const float* r = p + (size_t)(row0 + m) * ld + k0;
  v8f lo = *(const v8f*)(r + 16 * hf);
  v8f hi = *(const v8f*)(r + 16 * hf + 8);
  v16h a;
#pragma unroll
  for (int i = 0; i < 8; ++i) { a[i] = (_Float16)lo[i]; a[i + 8] = (_Float16)hi[i]; }
  return a;
}

// B fragment (32x16, f16) where B[k][n] = p[n][k] (p holds B transposed,
// contiguous along k): lane L -> n = n0 + (L&15), k = k0 + 16*half + 0..15.
__device__ __forceinline__ v16h load_b_T(const _Float16* p, int ld, int n0, int k0) {
  int lane = threadIdx.x & 31, nn = lane & 15, hf = lane >> 4;
  return *(const v16h*)(p + (size_t)(n0 + nn) * ld + k0 + 16 * hf);
}

// C/D fragment stores. VGPR r holds (row = row0 + r + 8*half, col = col0 + (L&15)).
__device__ __forceinline__ void store_c_f16(_Float16* p, int ld, int row0, int col0, v8f c) {
  int lane = threadIdx.x & 31, nn = lane & 15, hf = lane >> 4;
  _Float16* base = p + col0 + nn;
#pragma unroll
  for (int r = 0; r < 8; ++r) base[(size_t)(row0 + r + 8 * hf) * ld] = (_Float16)c[r];
}
// Transposed store: 8 contiguous f16 per lane -> single 16B store.
__device__ __forceinline__ void store_c_f16_T(_Float16* p, int ldT, int row0, int col0, v8f c) {
  int lane = threadIdx.x & 31, nn = lane & 15, hf = lane >> 4;
  v8h hv;
#pragma unroll
  for (int r = 0; r < 8; ++r) hv[r] = (_Float16)c[r];
  *(v8h*)(p + (size_t)(col0 + nn) * ldT + row0 + 8 * hf) = hv;
}
__device__ __forceinline__ void store_c_f32(float* p, int ld, int row0, int col0, v8f c) {
  int lane = threadIdx.x & 31, nn = lane & 15, hf = lane >> 4;
  float* base = p + col0 + nn;
#pragma unroll
  for (int r = 0; r < 8; ++r) base[(size_t)(row0 + r + 8 * hf) * ld] = c[r];
}

__device__ __forceinline__ float wave_reduce_sum(float v) {
#pragma unroll
  for (int off = 16; off > 0; off >>= 1) v += __shfl_xor(v, off, 32);
  return v;
}
__device__ __forceinline__ float wave_reduce_max(float v) {
#pragma unroll
  for (int off = 16; off > 0; off >>= 1) v = fmaxf(v, __shfl_xor(v, off, 32));
  return v;
}

// --------------------- K0: transpose + f32->f16 convert --------------------
// src [R][C] f32 (row-major) -> dst [C][R] f16.  block (32,8)
__global__ void transpose_cvt(const float* __restrict__ src, _Float16* __restrict__ dst,
                              int R, int C) {
  __shared__ float tile[32][33];
  int c0 = blockIdx.x * 32, r0 = blockIdx.y * 32;
  int tx = threadIdx.x, ty = threadIdx.y;
#pragma unroll
  for (int j = 0; j < 32; j += 8)
    tile[ty + j][tx] = src[(size_t)(r0 + ty + j) * C + c0 + tx];
  __syncthreads();
#pragma unroll
  for (int j = 0; j < 32; j += 8)
    dst[(size_t)(c0 + ty + j) * R + r0 + tx] = (_Float16)tile[tx][ty + j];
}

// ------------------- K1: QKV projection (x @ W per head) -------------------
// Wt: [H][DK][D] f16 (transposed weight). grid (N/32, H), block 128 (4 waves).
// outRM: [H][N][DK] f16 row-major (nullable). outT: [H][DK][N] f16 (nullable).
__global__ void qkv_proj(const float* __restrict__ x, const _Float16* __restrict__ Wt,
                         _Float16* __restrict__ outRM, _Float16* __restrict__ outT) {
  int n0 = blockIdx.x * 32;
  int h  = blockIdx.y;
  int wave = threadIdx.x >> 5;
  const _Float16* Wth = Wt + (size_t)h * kDK * kD;
  v8f acc[2][2] = {};
  for (int kc = 0; kc < kD / 32; ++kc) {
    v16h a0 = load_a_f32cvt(x, kD, n0, kc * 32);
    v16h a1 = load_a_f32cvt(x, kD, n0 + 16, kc * 32);
#pragma unroll
    for (int nt = 0; nt < 2; ++nt) {
      v16h b = load_b_T(Wth, kD, wave * 32 + nt * 16, kc * 32);
      acc[0][nt] = wmma_f16(a0, b, acc[0][nt]);
      acc[1][nt] = wmma_f16(a1, b, acc[1][nt]);
    }
  }
#pragma unroll
  for (int mt = 0; mt < 2; ++mt)
#pragma unroll
    for (int nt = 0; nt < 2; ++nt) {
      int col0 = wave * 32 + nt * 16;
      if (outRM) store_c_f16(outRM + (size_t)h * kN * kDK, kDK, n0 + mt * 16, col0, acc[mt][nt]);
      if (outT)  store_c_f16_T(outT + (size_t)h * kDK * kN, kN, n0 + mt * 16, col0, acc[mt][nt]);
    }
}

// ---------------- K2: Gram matrix G = K^T K and ksum (per head) ------------
// Kt: [H][DK][N] f16. G: [H][DK][DK] f32. ksum: [H][DK] f32. grid H, block 256.
__global__ void gram_kernel(const _Float16* __restrict__ Kt, float* __restrict__ G,
                            float* __restrict__ ksum) {
  int h = blockIdx.x;
  const _Float16* Kth = Kt + (size_t)h * kDK * kN;
  int wave = threadIdx.x >> 5;     // 8 waves, wave = row tile
  v8f acc[8] = {};
  for (int nc = 0; nc < kN / 32; ++nc) {
    v16h a = load_a_f16(Kth, kN, wave * 16, nc * 32);   // A[a][n] = Kt[a][n]
#pragma unroll
    for (int nt = 0; nt < 8; ++nt) {
      v16h b = load_b_T(Kth, kN, nt * 16, nc * 32);      // B[n][b] = Kt[b][n]
      acc[nt] = wmma_f16(a, b, acc[nt]);
    }
  }
  float* Gh = G + (size_t)h * kDK * kDK;
#pragma unroll
  for (int nt = 0; nt < 8; ++nt) store_c_f32(Gh, kDK, wave * 16, nt * 16, acc[nt]);
  // column sums of K (= row sums of Kt), vectorized 8-wide
  int t = threadIdx.x;
  if (t < kDK) {
    const v8h* row = (const v8h*)(Kth + (size_t)t * kN);
    float s = 0.f;
    for (int n = 0; n < kN / 8; ++n) {
      v8h v = row[n];
#pragma unroll
      for (int i = 0; i < 8; ++i) s += (float)v[i];
    }
    ksum[h * kDK + t] = s;
  }
}

// ------------- K3: closed-form row statistics mu[m], 1/sd[m] ---------------
// mu[m] = (Q[m]·ksum)/(N*sqrt(DK)); var = (QGQ/DK - N mu^2)/(N-1), ddof=1.
// grid (H, N/8), block 256 (8 waves, one row per wave).
__global__ void stats_kernel(const _Float16* __restrict__ Qh, const float* __restrict__ G,
                             const float* __restrict__ ksum, float* __restrict__ mu,
                             float* __restrict__ invsd) {
  int h = blockIdx.x;
  int wave = threadIdx.x >> 5, lane = threadIdx.x & 31;
  int m = blockIdx.y * 8 + wave;
  __shared__ float q[8][kDK];
  const _Float16* qr = Qh + ((size_t)h * kN + m) * kDK;
#pragma unroll
  for (int j = 0; j < 4; ++j) q[wave][lane + 32 * j] = (float)qr[lane + 32 * j];
  __syncthreads();
  const float* Gh = G + (size_t)h * kDK * kDK;
  const float* ks = ksum + h * kDK;
  float qGq = 0.f, qk = 0.f;
#pragma unroll
  for (int j = 0; j < 4; ++j) {
    int b = lane + 32 * j;
    float t = 0.f;
    for (int a = 0; a < kDK; ++a) t += q[wave][a] * Gh[a * kDK + b];
    qGq += t * q[wave][b];
    qk  += q[wave][b] * ks[b];
  }
  qGq = wave_reduce_sum(qGq);
  qk  = wave_reduce_sum(qk);
  if (lane == 0) {
    const float inv_sqrt_dk = 0.08838834764831845f;   // 1/sqrt(128)
    float mu_m   = qk * inv_sqrt_dk / (float)kN;
    float sum_s2 = qGq / (float)kDK;                  // sum_j s[m,j]^2
    float var = (sum_s2 - (float)kN * mu_m * mu_m) / (float)(kN - 1);
    mu[h * kN + m]    = mu_m;
    invsd[h * kN + m] = rsqrtf(fmaxf(var, 1e-20f));
  }
}

// ------------------------- K4: sigma = clamp(x @ Ws) -----------------------
// grid N, block 256 (8 waves = 8 heads). sig: [H][N] f32.
__global__ void sigma_kernel(const float* __restrict__ x, const float* __restrict__ Ws,
                             float* __restrict__ sig) {
  int n = blockIdx.x;
  __shared__ float xr[kD];
  for (int i = threadIdx.x; i < kD; i += 256) xr[i] = x[(size_t)n * kD + i];
  __syncthreads();
  int h = threadIdx.x >> 5, lane = threadIdx.x & 31;
  const float* w = Ws + (size_t)h * kD;
  float s = 0.f;
  for (int j = 0; j < kD / 32; ++j) s += xr[lane + 32 * j] * w[lane + 32 * j];
  s = wave_reduce_sum(s);
  if (lane == 0) sig[h * kN + n] = fmaxf(s, 0.001f);
}

// --------- K5: fused scores -> normalize -> softmax -> S + Z = S V ---------
// grid (H, N/16), block 128 (4 waves). LDS: 16 x (2048+8) f32 = 128.5 KB
// (CDNA5 WGP has 320 KB LDS -> 2 workgroups per WGP).
constexpr int kROWS = 16;
constexpr int kLD   = kN + 8;       // +8 pad: spreads rows across LDS banks
__global__ void attn_kernel(const _Float16* __restrict__ Qh, const _Float16* __restrict__ Kh,
                            const _Float16* __restrict__ Vt, const float* __restrict__ mu,
                            const float* __restrict__ invsd, float* __restrict__ Sout,
                            _Float16* __restrict__ Zh) {
  __shared__ float sc[kROWS * kLD];
  int h = blockIdx.x, n0 = blockIdx.y * kROWS;
  int wave = threadIdx.x >> 5, lane = threadIdx.x & 31;
  int nL = lane & 15, hf = lane >> 4;
  const _Float16* Qhh = Qh + (size_t)h * kN * kDK;
  const _Float16* Khh = Kh + (size_t)h * kN * kDK;
  const _Float16* Vth = Vt + (size_t)h * kDK * kN;
  const float inv_sqrt_dk = 0.08838834764831845f;

  // Q row-block A fragments, reused across all 128 column tiles.
  v16h aq[4];
#pragma unroll
  for (int kc = 0; kc < 4; ++kc) aq[kc] = load_a_f16(Qhh, kDK, n0, kc * 32);

  // Phase 1: scores tile (16 x N), normalized by per-column (mu, 1/sd).
  for (int ct = wave; ct < kN / 16; ct += 4) {
    v8f acc = {};
#pragma unroll
    for (int kc = 0; kc < 4; ++kc) {
      v16h b = load_b_T(Khh, kDK, ct * 16, kc * 32);   // B[k][m] = K[m][k]
      acc = wmma_f16(aq[kc], b, acc);
    }
    int col = ct * 16 + nL;
    float mum = mu[h * kN + col], isd = invsd[h * kN + col];
#pragma unroll
    for (int r = 0; r < 8; ++r) {
      float s = acc[r] * inv_sqrt_dk;
      sc[(r + 8 * hf) * kLD + col] = (s - mum) * isd;
    }
  }
  __syncthreads();

  // Phase 2: row softmax; write S (128-bit per-lane stores) and keep
  // normalized S in LDS for the Z GEMM.
  float* Sg = Sout + ((size_t)h * kN + n0) * kN;
  for (int row = wave * 4; row < wave * 4 + 4; ++row) {
    float* r = sc + row * kLD;
    float mx = -3.4e38f;
#pragma unroll 4
    for (int j = 0; j < kN / 128; ++j) {
      v4f v = *(const v4f*)(r + lane * 4 + 128 * j);
      mx = fmaxf(mx, fmaxf(fmaxf(v[0], v[1]), fmaxf(v[2], v[3])));
    }
    mx = wave_reduce_max(mx);
    float sum = 0.f;
#pragma unroll 4
    for (int j = 0; j < kN / 128; ++j) {
      v4f v = *(const v4f*)(r + lane * 4 + 128 * j);
      v4f e;
#pragma unroll
      for (int c = 0; c < 4; ++c) { e[c] = __expf(v[c] - mx); sum += e[c]; }
      *(v4f*)(r + lane * 4 + 128 * j) = e;
    }
    sum = wave_reduce_sum(sum);
    float inv = 1.0f / sum;
    float* srow = Sg + (size_t)row * kN;
#pragma unroll 4
    for (int j = 0; j < kN / 128; ++j) {
      v4f v = *(const v4f*)(r + lane * 4 + 128 * j);
      v4f p;
#pragma unroll
      for (int c = 0; c < 4; ++c) p[c] = v[c] * inv;
      *(v4f*)(r + lane * 4 + 128 * j) = p;              // reused by Z GEMM
      *(v4f*)(srow + lane * 4 + 128 * j) = p;           // coalesced b128 stores
    }
  }
  __syncthreads();

  // Phase 3: Z = S * V  (K-dim = N). A from LDS (f32->f16), B from Vt.
  v8f zacc[2] = {};
  for (int mc = 0; mc < kN / 32; ++mc) {
    v16h a = load_a_f32cvt(sc, kLD, 0, mc * 32);
#pragma unroll
    for (int nt = 0; nt < 2; ++nt) {
      v16h b = load_b_T(Vth, kN, wave * 32 + nt * 16, mc * 32);  // B[m][k]=Vt[k][m]
      zacc[nt] = wmma_f16(a, b, zacc[nt]);
    }
  }
  _Float16* Zhh = Zh + (size_t)h * kN * kDK;
#pragma unroll
  for (int nt = 0; nt < 2; ++nt)
    store_c_f16(Zhh, kDK, n0, wave * 32 + nt * 16, zacc[nt]);
}

// ------------------- K6: out = concat(Z) @ W0 + b0 -------------------------
// grid (N/32, D/64), block 128 (4 waves, 16 cols each).
__global__ void out_proj(const _Float16* __restrict__ Zh, const _Float16* __restrict__ W0t,
                         const float* __restrict__ b0, float* __restrict__ outp) {
  int n0 = blockIdx.x * 32;
  int j0 = blockIdx.y * 64 + (threadIdx.x >> 5) * 16;
  int lane = threadIdx.x & 31, nL = lane & 15, hf = lane >> 4;
  v8f acc[2] = {};
  for (int kc = 0; kc < kD / 32; ++kc) {
    int c = kc * 32;
    const _Float16* Zhh = Zh + (size_t)(c >> 7) * kN * kDK;  // head = c/128
    v16h a0 = load_a_f16(Zhh, kDK, n0, c & 127);
    v16h a1 = load_a_f16(Zhh, kDK, n0 + 16, c & 127);
    v16h b  = load_b_T(W0t, kD, j0, c);
    acc[0] = wmma_f16(a0, b, acc[0]);
    acc[1] = wmma_f16(a1, b, acc[1]);
  }
  float bb = b0[j0 + nL];
#pragma unroll
  for (int mt = 0; mt < 2; ++mt)
#pragma unroll
    for (int r = 0; r < 8; ++r)
      outp[(size_t)(n0 + mt * 16 + r + 8 * hf) * kD + j0 + nL] = acc[mt][r] + bb;
}

// --------------------- K7: prior association P -----------------------------
// P[h,n,m] = exp(-0.5((n-m)/sigma)^2) / rowsum  (1/sqrt(2*pi*sigma) cancels).
// grid (H, N/8), block 256 (8 waves, one row each). 64 KB LDS.
__global__ void prior_kernel(const float* __restrict__ sig, float* __restrict__ P) {
  int h = blockIdx.x, wave = threadIdx.x >> 5, lane = threadIdx.x & 31;
  int n = blockIdx.y * 8 + wave;
  __shared__ float buf[8][kN];
  float isg = 1.0f / sig[h * kN + n];
  float sum = 0.f;
#pragma unroll 4
  for (int j = 0; j < kN / 128; ++j) {
    int m0 = lane * 4 + 128 * j;
    v4f e;
#pragma unroll
    for (int c = 0; c < 4; ++c) {
      float d = (float)(n - (m0 + c)) * isg;
      e[c] = __expf(-0.5f * d * d);
      sum += e[c];
    }
    *(v4f*)(&buf[wave][m0]) = e;
  }
  sum = wave_reduce_sum(sum);
  float inv = 1.0f / sum;
  float* pr = P + ((size_t)h * kN + n) * kN;
#pragma unroll 4
  for (int j = 0; j < kN / 128; ++j) {
    int m0 = lane * 4 + 128 * j;
    v4f v = *(const v4f*)(&buf[wave][m0]);
    v4f p;
#pragma unroll
    for (int c = 0; c < 4; ++c) p[c] = v[c] * inv;
    *(v4f*)(pr + m0) = p;                                // coalesced b128 stores
  }
}

// ---------------------------------------------------------------------------

extern "C" void kernel_launch(void* const* d_in, const int* in_sizes, int n_in,
                              void* d_out, int out_size, void* d_ws, size_t ws_size,
                              hipStream_t stream) {
  const float* x  = (const float*)d_in[0];
  const float* Wq = (const float*)d_in[1];
  const float* Wk = (const float*)d_in[2];
  const float* Wv = (const float*)d_in[3];
  const float* Ws = (const float*)d_in[4];
  const float* W0 = (const float*)d_in[5];
  const float* b0 = (const float*)d_in[6];

  float* out = (float*)d_out;                     // [N, D]
  float* P   = out + (size_t)kN * kD;             // [H, N, N]
  float* S   = P + (size_t)kH * kN * kN;          // [H, N, N]

  // Workspace carve-out (~29 MB total).
  char* w = (char*)d_ws;
  auto carve = [&](size_t bytes) -> void* {
    void* p = (void*)w;
    w += (bytes + 255) & ~(size_t)255;
    return p;
  };
  _Float16* Wqt = (_Float16*)carve(sizeof(_Float16) * kH * kDK * kD);
  _Float16* Wkt = (_Float16*)carve(sizeof(_Float16) * kH * kDK * kD);
  _Float16* Wvt = (_Float16*)carve(sizeof(_Float16) * kH * kDK * kD);
  _Float16* W0t = (_Float16*)carve(sizeof(_Float16) * kD * kD);
  _Float16* Qh  = (_Float16*)carve(sizeof(_Float16) * kH * kN * kDK);
  _Float16* Kh  = (_Float16*)carve(sizeof(_Float16) * kH * kN * kDK);
  _Float16* Kt  = (_Float16*)carve(sizeof(_Float16) * kH * kDK * kN);
  _Float16* Vt  = (_Float16*)carve(sizeof(_Float16) * kH * kDK * kN);
  _Float16* Zh  = (_Float16*)carve(sizeof(_Float16) * kH * kN * kDK);
  float* G      = (float*)carve(sizeof(float) * kH * kDK * kDK);
  float* ksum   = (float*)carve(sizeof(float) * kH * kDK);
  float* muv    = (float*)carve(sizeof(float) * kH * kN);
  float* invsd  = (float*)carve(sizeof(float) * kH * kN);
  float* sig    = (float*)carve(sizeof(float) * kH * kN);

  dim3 tb(32, 8);
  for (int h = 0; h < kH; ++h) {
    transpose_cvt<<<dim3(kDK / 32, kD / 32), tb, 0, stream>>>(
        Wq + (size_t)h * kD * kDK, Wqt + (size_t)h * kDK * kD, kD, kDK);
    transpose_cvt<<<dim3(kDK / 32, kD / 32), tb, 0, stream>>>(
        Wk + (size_t)h * kD * kDK, Wkt + (size_t)h * kDK * kD, kD, kDK);
    transpose_cvt<<<dim3(kDK / 32, kD / 32), tb, 0, stream>>>(
        Wv + (size_t)h * kD * kDK, Wvt + (size_t)h * kDK * kD, kD, kDK);
  }
  transpose_cvt<<<dim3(kD / 32, kD / 32), tb, 0, stream>>>(W0, W0t, kD, kD);

  qkv_proj<<<dim3(kN / 32, kH), 128, 0, stream>>>(x, Wqt, Qh, (_Float16*)nullptr);
  qkv_proj<<<dim3(kN / 32, kH), 128, 0, stream>>>(x, Wkt, Kh, Kt);
  qkv_proj<<<dim3(kN / 32, kH), 128, 0, stream>>>(x, Wvt, (_Float16*)nullptr, Vt);

  gram_kernel<<<kH, 256, 0, stream>>>(Kt, G, ksum);
  stats_kernel<<<dim3(kH, kN / 8), 256, 0, stream>>>(Qh, G, ksum, muv, invsd);
  sigma_kernel<<<kN, 256, 0, stream>>>(x, Ws, sig);

  attn_kernel<<<dim3(kH, kN / kROWS), 128, 0, stream>>>(Qh, Kh, Vt, muv, invsd, S, Zh);
  out_proj<<<dim3(kN / 32, kD / 64), 128, 0, stream>>>(Zh, W0t, b0, out);
  prior_kernel<<<dim3(kH, kN / 8), 256, 0, stream>>>(sig, P);
}
